// ScaledDotProductAttention_45088566673887
// MI455X (gfx1250) — matrix operand order repaired and checked by reading, not compiled, not run on hardware
//
#include <hip/hip_runtime.h>

// ---------------------------------------------------------------------------
// Flash-attention (causal + key-padding mask, keras-style EPS normalization)
// B=8, SQ=SK=2048, DK=DV=256, fp32 in/out, f16 WMMA math w/ f32 accumulate.
// K/V tiles staged via CDNA5 async-to-LDS DMA, overlapped with WMMA compute.
// ---------------------------------------------------------------------------

typedef __attribute__((ext_vector_type(16))) _Float16 v16h;
typedef __attribute__((ext_vector_type(8)))  float    v8f;
typedef __attribute__((ext_vector_type(4)))  _Float16 h4;

#define BATCH 8
#define SQ    2048
#define SK    2048
#define DK    256
#define DV    256
#define EPS   1e-7f

// f16 tiles: K [key][dk] row-major, V transposed [dv][key], P per-wave.
// Row paddings keep every fragment read a 16B-aligned ds_load_b128.
#define KPAD  (DK + 8)    // 264 halves -> 528B rows (16B aligned)
#define VPAD  (32 + 8)    // 40 halves  -> 80B rows  (16B aligned)
#define PPAD  48          // 96B rows   (16B aligned)

__device__ __forceinline__ void cvt8(v16h& f, int base, float4 a, float4 b, float s) {
    f[base + 0] = (_Float16)(a.x * s); f[base + 1] = (_Float16)(a.y * s);
    f[base + 2] = (_Float16)(a.z * s); f[base + 3] = (_Float16)(a.w * s);
    f[base + 4] = (_Float16)(b.x * s); f[base + 5] = (_Float16)(b.y * s);
    f[base + 6] = (_Float16)(b.z * s); f[base + 7] = (_Float16)(b.w * s);
}

__launch_bounds__(128, 1)
__global__ void sdpa_flash_wmma(const float* __restrict__ Q,
                                const float* __restrict__ K,
                                const float* __restrict__ V,
                                const unsigned char* __restrict__ mask,
                                float* __restrict__ Out)
{
    // Raw fp32 staging filled by the async-to-LDS engine (ASYNCcnt domain).
    __shared__ float   Ks[32 * DK];          // 32 KB
    __shared__ float   Vs[32 * DV];          // 32 KB
    __shared__ _Float16 Klds[32][KPAD];      // 16.5 KB  [key][dk]
    __shared__ _Float16 Vlds[DV][VPAD];      // 20.0 KB  [dv][key] (transposed)
    __shared__ _Float16 Plds[4][16][PPAD];   //  6.0 KB  [wave][m][k]

    const int tid  = threadIdx.x;
    const int w    = tid >> 5;          // wave id 0..3
    const int lane = tid & 31;
    const int n    = lane & 15;         // N column (C/D, B layouts)
    const int hi   = lane >> 4;         // lane half: +8 row (C), K group (A/B)
    const int qb   = blockIdx.x;        // 64-row q block
    const int b    = blockIdx.y;

    const int qbase = qb * 64 + w * 16; // first q row of this wave's tile

    const int nblocks = 2 * qb + 2;     // causal: keys 0 .. qb*64+63
    const float*         Kb = K    + (size_t)b * SK * DK;
    const float*         Vb = V    + (size_t)b * SK * DV;
    const unsigned char* mb = mask + (size_t)b * SK;

    // ---- Load Q tile as persistent A-fragments, pre-scaled by 1/sqrt(DK) ----
    // A-layout (16x32 f16): lane m = lane%16, group g = lane>>4:
    //   halves 0..7  = K(dk) = c*32 +      g*8 .. +7
    //   halves 8..15 = K(dk) = c*32 + 16 + g*8 .. +7
    v16h qf[8];
    {
        const float sc = 0.0625f;  // 1/16
        const float* qrow = Q + ((size_t)b * SQ + qbase + n) * DK;
        #pragma unroll
        for (int c = 0; c < 8; ++c) {
            float4 a0 = *(const float4*)(qrow + c * 32 + hi * 8);
            float4 a1 = *(const float4*)(qrow + c * 32 + hi * 8 + 4);
            float4 b0 = *(const float4*)(qrow + c * 32 + 16 + hi * 8);
            float4 b1 = *(const float4*)(qrow + c * 32 + 16 + hi * 8 + 4);
            v16h f;
            cvt8(f, 0, a0, a1, sc);
            cvt8(f, 8, b0, b1, sc);
            qf[c] = f;
        }
    }

    v8f acc[16];                 // 16x256 output accumulator (16 DV chunks)
    const v8f vzero = {0.f,0.f,0.f,0.f,0.f,0.f,0.f,0.f};
    #pragma unroll
    for (int t = 0; t < 16; ++t) acc[t] = vzero;

    float mrun[8], lrun[8];      // per-row running max / sum (row = r + 8*hi)
    #pragma unroll
    for (int r = 0; r < 8; ++r) { mrun[r] = -1e30f; lrun[r] = 0.f; }

    // Async DMA of one 32-key K+V tile pair into fp32 staging.
    // Per wave: 32 x b128 ops x 32 lanes = 16 KB; 4 waves cover 64 KB.
    auto issue_async = [&](int j) {
        const int jbase = j * 32;
        const float* Kg = Kb + (size_t)jbase * DK;
        const float* Vg = Vb + (size_t)jbase * DV;
        #pragma unroll
        for (int i = 0; i < 16; ++i) {
            const int idx = tid + i * 128;               // float4 idx 0..2047
            unsigned     koff = (unsigned)(size_t)&Ks[idx * 4];
            unsigned     voff = (unsigned)(size_t)&Vs[idx * 4];
            const float* kg   = Kg + (size_t)idx * 4;
            const float* vg   = Vg + (size_t)idx * 4;
            asm volatile("global_load_async_to_lds_b128 %0, %1, off"
                         :: "v"(koff), "v"(kg) : "memory");
            asm volatile("global_load_async_to_lds_b128 %0, %1, off"
                         :: "v"(voff), "v"(vg) : "memory");
        }
    };

    issue_async(0);   // prologue: stage first tile

    for (int j = 0; j < nblocks; ++j) {
        const int jbase = j * 32;

        // Drain this wave's async copies, then publish across the WG.
        asm volatile("s_wait_asynccnt 0x0" ::: "memory");
        __syncthreads();

        // ---- Convert staging fp32 -> f16 tiles (K natural, V transposed) ----
        #pragma unroll
        for (int i = 0; i < 16; ++i) {
            const int idx = tid + i * 128;
            const int key = idx >> 6;            // 64 float4 per 256-wide row
            const int col = (idx & 63) << 2;
            float4 kv = *(const float4*)&Ks[idx * 4];
            h4 kh = { (_Float16)kv.x, (_Float16)kv.y,
                      (_Float16)kv.z, (_Float16)kv.w };
            *(h4*)&Klds[key][col] = kh;
            float4 vv = *(const float4*)&Vs[idx * 4];
            Vlds[col + 0][key] = (_Float16)vv.x;   // transpose on store
            Vlds[col + 1][key] = (_Float16)vv.y;
            Vlds[col + 2][key] = (_Float16)vv.z;
            Vlds[col + 3][key] = (_Float16)vv.w;
        }
        __syncthreads();   // tiles ready; staging fully consumed by all waves

        // Prefetch next tile into staging, overlapping all compute below.
        if (j + 1 < nblocks) issue_async(j + 1);

        // ---- S = (Q/16) . K^T : two 16x16 f32 tiles (keys n, n+16) ----
        // B-layout (32x16 f16): lane N = lane%16, K group g = lane>>4,
        // K(dk) = g*8..+7 and 16+g*8..+7 per lane -> 2x ds_load_b128.
        v8f S0 = vzero, S1 = vzero;
        #pragma unroll
        for (int c = 0; c < 8; ++c) {
            union { v16h h; uint4 u[2]; } b0, b1;
            const _Float16* k0 = &Klds[n][c * 32 + hi * 8];
            b0.u[0] = *(const uint4*)k0;
            b0.u[1] = *(const uint4*)(k0 + 16);
            const _Float16* k1 = &Klds[16 + n][c * 32 + hi * 8];
            b1.u[0] = *(const uint4*)k1;
            b1.u[1] = *(const uint4*)(k1 + 16);
            S0 = __builtin_amdgcn_wmma_f32_16x16x32_f16(false, qf[c], false, b0.h,
                                                        (short)0, S0, false, false);
            S1 = __builtin_amdgcn_wmma_f32_16x16x32_f16(false, qf[c], false, b1.h,
                                                        (short)0, S1, false, false);
        }

        // ---- Online softmax. Reference takes max over RAW scores, then
        // masks probabilities; we do the same within processed blocks. ----
        const float mk0 = mb[jbase + n]      ? 1.f : 0.f;
        const float mk1 = mb[jbase + 16 + n] ? 1.f : 0.f;
        float corr[8];
        #pragma unroll
        for (int r = 0; r < 8; ++r) {
            float s0 = S0[r], s1 = S1[r];
            float mx = fmaxf(s0, s1);
            #pragma unroll
            for (int off = 1; off < 16; off <<= 1)   // reduce within 16-lane half
                mx = fmaxf(mx, __shfl_xor(mx, off, 32));
            const float mnew = fmaxf(mrun[r], mx);
            const int qrow = qbase + r + 8 * hi;
            float p0 = __expf(s0 - mnew);
            float p1 = __expf(s1 - mnew);
            p0 = (jbase + n      > qrow) ? 0.f : p0 * mk0;   // causal + padding
            p1 = (jbase + 16 + n > qrow) ? 0.f : p1 * mk1;
            const float c0 = __expf(mrun[r] - mnew);
            float rs = p0 + p1;
            #pragma unroll
            for (int off = 1; off < 16; off <<= 1)
                rs += __shfl_xor(rs, off, 32);
            lrun[r] = lrun[r] * c0 + rs;
            mrun[r] = mnew;
            corr[r] = c0;
            const int m = r + 8 * hi;        // C-layout row
            Plds[w][m][n]      = (_Float16)p0;
            Plds[w][m][16 + n] = (_Float16)p1;
        }
        #pragma unroll
        for (int t = 0; t < 16; ++t) {
            #pragma unroll
            for (int r = 0; r < 8; ++r) acc[t][r] *= corr[r];
        }

        // Per-wave LDS round trip (C-layout -> A-layout). LDS ops are in-order
        // within a wave; only the DS counter needs draining (ASYNCcnt is a
        // separate domain), no WG barrier required here.
        asm volatile("s_wait_dscnt 0x0" ::: "memory");

        union { v16h h; uint4 u[2]; } pa;
        pa.u[0] = *(const uint4*)&Plds[w][n][hi * 8];        // K = g*8..+7
        pa.u[1] = *(const uint4*)&Plds[w][n][16 + hi * 8];   // K = 16+g*8..+7

        // ---- O += P . V  (V read from transposed LDS tile) ----
        #pragma unroll
        for (int t = 0; t < 16; ++t) {
            union { v16h h; uint4 u[2]; } vb;
            const _Float16* vp = &Vlds[t * 16 + n][hi * 8];
            vb.u[0] = *(const uint4*)vp;
            vb.u[1] = *(const uint4*)(vp + 16);
            acc[t] = __builtin_amdgcn_wmma_f32_16x16x32_f16(false, pa.h, false, vb.h,
                                                            (short)0, acc[t], false, false);
        }
        __syncthreads();   // f16 tiles free for next block's convert pass
    }

    // ---- Normalize by (sum + EPS) and store ----
    float inv[8];
    #pragma unroll
    for (int r = 0; r < 8; ++r) inv[r] = 1.f / (lrun[r] + EPS);

    float* ob = Out + ((size_t)b * SQ + qbase) * DV;
    #pragma unroll
    for (int r = 0; r < 8; ++r) {
        const int m = r + 8 * hi;
        float* orow = ob + (size_t)m * DV + n;
        #pragma unroll
        for (int t = 0; t < 16; ++t)
            orow[t * 16] = acc[t][r] * inv[r];
    }
}

extern "C" void kernel_launch(void* const* d_in, const int* in_sizes, int n_in,
                              void* d_out, int out_size, void* d_ws, size_t ws_size,
                              hipStream_t stream) {
    const float*         Qp = (const float*)d_in[0];
    const float*         Kp = (const float*)d_in[1];
    const float*         Vp = (const float*)d_in[2];
    const unsigned char* Mp = (const unsigned char*)d_in[3];  // jax bool = 1 byte
    float* Op = (float*)d_out;

    dim3 grid(SQ / 64, BATCH);   // 32 q-blocks x 8 batches = 256 WGs
    dim3 block(128);             // 4 waves, one 16-row q tile each
    sdpa_flash_wmma<<<grid, block, 0, stream>>>(Qp, Kp, Vp, Mp, Op);
}
